// Learner_77910706749781
// MI455X (gfx1250) — matrix operand order repaired and checked by reading, not compiled, not run on hardware
//
#include <hip/hip_runtime.h>
#include <hip/hip_bf16.h>
#include <math.h>

typedef __attribute__((ext_vector_type(16))) __bf16 v16bf;
typedef __attribute__((ext_vector_type(8)))  float  v8f;
typedef __attribute__((ext_vector_type(4)))  int    v4i;

#define WMMA_BF16(A, B, C) \
  __builtin_amdgcn_wmma_f32_16x16x32_bf16(false, (A), false, (B), (short)0, (C), false, false)

#if defined(__AMDGCN__) && __has_builtin(__builtin_amdgcn_global_load_async_to_lds_b128)
#define USE_ASYNC_LDS 1
#else
#define USE_ASYNC_LDS 0
#endif

__device__ __forceinline__ void waitAsyncLds() {
#if USE_ASYNC_LDS
#if __has_builtin(__builtin_amdgcn_s_wait_asynccnt)
  __builtin_amdgcn_s_wait_asynccnt(0);
#else
  asm volatile("s_wait_asynccnt 0" ::: "memory");
#endif
#endif
}

// ---------------- problem constants ----------------
#define N_ROWS 16384
#define DIM    2048
#define H1     512
#define H2     32
#define MROWS  2048

// prep element counts (bf16 elements in swizzled layouts)
#define NW1E (64 * 32 * 32 * 16)   // [kchunk64][tile32][lane32][16]
#define NW2E (16 * 2 * 32 * 16)    // [kchunk16][tile2][lane32][16]
#define NME  (128 * 32 * 16)       // [tile128][lane32][16]
#define NPREP (NW1E + NW2E + NME + NME)

// ---------------- workspace layout (bytes) ----------------
#define OFF_W1H  ((size_t)0)
#define OFF_W1L  (OFF_W1H + (size_t)NW1E * 2)
#define OFF_W2H  (OFF_W1L + (size_t)NW1E * 2)
#define OFF_W2L  (OFF_W2H + (size_t)NW2E * 2)
#define OFF_AMH  (OFF_W2L + (size_t)NW2E * 2)
#define OFF_AML  (OFF_AMH + (size_t)NME * 2)
#define OFF_NMH  (OFF_AML + (size_t)NME * 2)
#define OFF_NML  (OFF_NMH + (size_t)NME * 2)
#define OFF_H    (OFF_NML + (size_t)NME * 2)   // fp32 hidden [16384][512]

// split fp32 -> bf16 hi + bf16 residual, packed in WMMA A/B register order
__device__ __forceinline__ void splitToBf16(const float* f, v16bf& hi, v16bf& lo) {
#pragma unroll
  for (int i = 0; i < 16; ++i) {
    float v = f[i];
    __bf16 h = (__bf16)v;
    hi[i] = h;
    lo[i] = (__bf16)(v - (float)h);
  }
}

// ================= K0: convert + swizzle weights/memories =================
__global__ __launch_bounds__(256) void prep_kernel(
    const float* __restrict__ W1, const float* __restrict__ W2,
    const float* __restrict__ am, const float* __restrict__ nm,
    __bf16* __restrict__ W1h, __bf16* __restrict__ W1l,
    __bf16* __restrict__ W2h, __bf16* __restrict__ W2l,
    __bf16* __restrict__ amh, __bf16* __restrict__ aml,
    __bf16* __restrict__ nmh, __bf16* __restrict__ nml) {
  int idx = blockIdx.x * blockDim.x + threadIdx.x;
  if (idx >= NPREP) return;
  float v;
  __bf16 *dh, *dl;
  int di;
  if (idx < NW1E) {
    // B-layout for GEMM1: row=T*16+(lane&15), K=c*32+8*(lane>>4)+e for e<8 ; K=c*32+16+8*(lane>>4)+(e-8) for e>=8
    int e = idx & 15, l = (idx >> 4) & 31, T = (idx >> 9) & 31, c = idx >> 14;
    int row = T * 16 + (l & 15);
    int k = c * 32 + ((e & 8) << 1) + ((l >> 4) << 3) + (e & 7);
    v = W1[(size_t)row * DIM + k];
    dh = W1h; dl = W1l; di = idx;
  } else if (idx < NW1E + NW2E) {
    int j = idx - NW1E;
    int e = j & 15, l = (j >> 4) & 31, tf = (j >> 9) & 1, cc = j >> 10;
    int m = tf * 16 + (l & 15);
    int k = cc * 32 + ((e & 8) << 1) + ((l >> 4) << 3) + (e & 7);
    v = W2[(size_t)m * H1 + k];
    dh = W2h; dl = W2l; di = j;
  } else if (idx < NW1E + NW2E + NME) {
    int j = idx - (NW1E + NW2E);
    int e = j & 15, l = (j >> 4) & 31, t = j >> 9;
    int row = t * 16 + (l & 15);
    int k = ((e & 8) << 1) + ((l >> 4) << 3) + (e & 7);     // K = feat dim (32), single chunk
    v = am[(size_t)row * H2 + k];
    dh = amh; dl = aml; di = j;
  } else {
    int j = idx - (NW1E + NW2E + NME);
    int e = j & 15, l = (j >> 4) & 31, t = j >> 9;
    int row = t * 16 + (l & 15);
    int k = ((e & 8) << 1) + ((l >> 4) << 3) + (e & 7);
    v = nm[(size_t)row * H2 + k];
    dh = nmh; dl = nml; di = j;
  }
  __bf16 h = (__bf16)v;
  dh[di] = h;
  dl[di] = (__bf16)(v - (float)h);
}

// ================= K1: h = relu(x @ W1^T + b1), split-bf16 WMMA =================
// grid: 256 blocks of 256 threads (8 waves). block covers 128 rows x 256 H1 cols.
__global__ __launch_bounds__(256) void gemm1_kernel(
    const float* __restrict__ x, const __bf16* __restrict__ W1h,
    const __bf16* __restrict__ W1l, const float* __restrict__ b1,
    float* __restrict__ hOut) {
  __shared__ __align__(32) uint4 sW[2][2][16][32][2];  // [buf][hi/lo][tile][lane][16B] = 64KB
  const int tid = threadIdx.x;
  const int w = tid >> 5, lane = tid & 31;
  const int sel = lane >> 4, l15 = lane & 15;
  const int colHalf = blockIdx.x & 1;
  const int rowBlock = blockIdx.x >> 1;
  const int g = rowBlock * 8 + w;            // 16-row group per wave

  const uint4* gWh = (const uint4*)W1h;
  const uint4* gWl = (const uint4*)W1l;

  // Stage one 32KB W1 hi/lo K-chunk into LDS.  On gfx1250 use the async
  // Global->LDS DMA path (no VGPR round trip, tracked by ASYNCcnt).
  auto loadChunk = [&](int c, int buf) {
#pragma unroll
    for (int i = 0; i < 8; ++i) {
      int idx = i * 256 + tid;               // 0..2047 uint4
      int hl = idx >> 10, rem = idx & 1023;
      int t = rem >> 6, r2 = rem & 63, ll = r2 >> 1, part = r2 & 1;
      const uint4* src = hl ? gWl : gWh;
      const uint4* sp = &src[((size_t)(c * 32 + colHalf * 16 + t) * 32 + ll) * 2 + part];
      uint4* dp = &sW[buf][hl][t][ll][part];
#if USE_ASYNC_LDS
      __builtin_amdgcn_global_load_async_to_lds_b128(
          (__attribute__((address_space(1))) v4i*)sp,
          (__attribute__((address_space(3))) v4i*)dp, 0, 0);
#else
      *dp = *sp;
#endif
    }
  };

  v8f C[16];
#pragma unroll
  for (int t = 0; t < 16; ++t)
#pragma unroll
    for (int r = 0; r < 8; ++r) C[t][r] = 0.0f;

  const int rowA = g * 16 + l15;
  const float* xrow = x + (size_t)rowA * DIM;

  loadChunk(0, 0);
  waitAsyncLds();
  __syncthreads();

  for (int c = 0; c < 64; ++c) {
    if (c + 1 < 64) {
      loadChunk(c + 1, (c + 1) & 1);
      __builtin_prefetch(xrow + (c + 1) * 32, 0, 0);   // global_prefetch_b8
    }
    // A operand: 16 fp32 from this lane's x row, in WMMA A register order
    const float* pk = xrow + c * 32 + sel * 8;
    float f[16];
    {
      float4 q0 = *(const float4*)(pk);
      float4 q1 = *(const float4*)(pk + 4);
      float4 q2 = *(const float4*)(pk + 16);
      float4 q3 = *(const float4*)(pk + 20);
      f[0]=q0.x; f[1]=q0.y; f[2]=q0.z; f[3]=q0.w;
      f[4]=q1.x; f[5]=q1.y; f[6]=q1.z; f[7]=q1.w;
      f[8]=q2.x; f[9]=q2.y; f[10]=q2.z; f[11]=q2.w;
      f[12]=q3.x; f[13]=q3.y; f[14]=q3.z; f[15]=q3.w;
    }
    v16bf ah, al;
    splitToBf16(f, ah, al);
    const int buf = c & 1;
#pragma unroll
    for (int t = 0; t < 16; ++t) {
      v16bf bh = *(const v16bf*)&sW[buf][0][t][lane][0];
      v16bf bl = *(const v16bf*)&sW[buf][1][t][lane][0];
      v8f acc = C[t];
      acc = WMMA_BF16(al, bh, acc);
      acc = WMMA_BF16(ah, bl, acc);
      acc = WMMA_BF16(ah, bh, acc);
      C[t] = acc;
    }
    waitAsyncLds();   // drain this wave's async stores before the barrier
    __syncthreads();
  }

  // epilogue: +b1, relu, store fp32 h
#pragma unroll
  for (int t = 0; t < 16; ++t) {
    int col = colHalf * 256 + t * 16 + l15;
    float bias = b1[col];
    int rbase = g * 16 + 8 * sel;
#pragma unroll
    for (int r = 0; r < 8; ++r) {
      float v = C[t][r] + bias;
      v = v > 0.0f ? v : 0.0f;
      hOut[(size_t)(rbase + r) * H1 + col] = v;
    }
  }
}

// ================= K2: feat = h@W2^T+b2 ; scores ; sigmoid =================
// grid: 128 blocks of 256 threads; each wave handles one 16-row group.
__global__ __launch_bounds__(256) void head_kernel(
    const float* __restrict__ hIn, const __bf16* __restrict__ W2h,
    const __bf16* __restrict__ W2l, const float* __restrict__ b2,
    const __bf16* __restrict__ amh, const __bf16* __restrict__ aml,
    const __bf16* __restrict__ nmh, const __bf16* __restrict__ nml,
    float* __restrict__ out) {
  __shared__ float sFeat[8][16][32];
  const int tid = threadIdx.x, w = tid >> 5, lane = tid & 31;
  const int sel = lane >> 4, l15 = lane & 15;
  const int g = blockIdx.x * 8 + w;

  v8f Cf[2];
#pragma unroll
  for (int tf = 0; tf < 2; ++tf)
#pragma unroll
    for (int r = 0; r < 8; ++r) Cf[tf][r] = 0.0f;

  const int rowA = g * 16 + l15;
  const float* hrow = hIn + (size_t)rowA * H1;

  for (int cc = 0; cc < 16; ++cc) {
    const float* pk = hrow + cc * 32 + sel * 8;
    float f[16];
    {
      float4 q0 = *(const float4*)(pk);
      float4 q1 = *(const float4*)(pk + 4);
      float4 q2 = *(const float4*)(pk + 16);
      float4 q3 = *(const float4*)(pk + 20);
      f[0]=q0.x; f[1]=q0.y; f[2]=q0.z; f[3]=q0.w;
      f[4]=q1.x; f[5]=q1.y; f[6]=q1.z; f[7]=q1.w;
      f[8]=q2.x; f[9]=q2.y; f[10]=q2.z; f[11]=q2.w;
      f[12]=q3.x; f[13]=q3.y; f[14]=q3.z; f[15]=q3.w;
    }
    v16bf ah, al;
    splitToBf16(f, ah, al);
#pragma unroll
    for (int tf = 0; tf < 2; ++tf) {
      v16bf bh = *(const v16bf*)(W2h + ((size_t)(cc * 2 + tf) * 32 + lane) * 16);
      v16bf bl = *(const v16bf*)(W2l + ((size_t)(cc * 2 + tf) * 32 + lane) * 16);
      v8f acc = Cf[tf];
      acc = WMMA_BF16(al, bh, acc);
      acc = WMMA_BF16(ah, bl, acc);
      acc = WMMA_BF16(ah, bh, acc);
      Cf[tf] = acc;
    }
  }

  // + b2 (bias per feat column == per lane within tile)
#pragma unroll
  for (int tf = 0; tf < 2; ++tf) {
    float bb = b2[tf * 16 + l15];
#pragma unroll
    for (int r = 0; r < 8; ++r) Cf[tf][r] += bb;
  }

  // transpose feat C-layout -> A-layout via LDS
#pragma unroll
  for (int tf = 0; tf < 2; ++tf)
#pragma unroll
    for (int r = 0; r < 8; ++r)
      sFeat[w][r + 8 * sel][tf * 16 + l15] = Cf[tf][r];
  __syncthreads();

  float ff[16];
  {
    const float* fr = &sFeat[w][l15][0];
#pragma unroll
    for (int i = 0; i < 8; ++i) {
      ff[i]     = fr[sel * 8 + i];
      ff[8 + i] = fr[16 + sel * 8 + i];
    }
  }
  v16bf fah, fal;
  splitToBf16(ff, fah, fal);

  float amax[8], nmax[8];
  for (int mem = 0; mem < 2; ++mem) {
    const __bf16* mh = mem ? nmh : amh;
    const __bf16* ml = mem ? nml : aml;
    v8f rmax;
#pragma unroll
    for (int r = 0; r < 8; ++r) rmax[r] = -INFINITY;
    for (int t = 0; t < 128; ++t) {
      v16bf bh = *(const v16bf*)(mh + ((size_t)t * 32 + lane) * 16);
      v16bf bl = *(const v16bf*)(ml + ((size_t)t * 32 + lane) * 16);
      v8f cz = {0.f, 0.f, 0.f, 0.f, 0.f, 0.f, 0.f, 0.f};
      cz = WMMA_BF16(fal, bh, cz);
      cz = WMMA_BF16(fah, bl, cz);
      cz = WMMA_BF16(fah, bh, cz);
#pragma unroll
      for (int r = 0; r < 8; ++r) rmax[r] = fmaxf(rmax[r], cz[r]);
    }
    // reduce max across the 16 lanes holding this row's columns (wave32 butterfly)
#pragma unroll
    for (int r = 0; r < 8; ++r) {
      float v = rmax[r];
      v = fmaxf(v, __shfl_xor(v, 1, 32));
      v = fmaxf(v, __shfl_xor(v, 2, 32));
      v = fmaxf(v, __shfl_xor(v, 4, 32));
      v = fmaxf(v, __shfl_xor(v, 8, 32));
      if (mem) nmax[r] = v; else amax[r] = v;
    }
  }

  if (l15 == 0) {
#pragma unroll
    for (int r = 0; r < 8; ++r) {
      float z = (amax[r] - nmax[r]) * (1.0f / 32.0f);
      out[g * 16 + r + 8 * sel] = 1.0f / (1.0f + expf(-z));
    }
  }
}

// ================= host =================
extern "C" void kernel_launch(void* const* d_in, const int* in_sizes, int n_in,
                              void* d_out, int out_size, void* d_ws, size_t ws_size,
                              hipStream_t stream) {
  const float* x  = (const float*)d_in[0];
  const float* W1 = (const float*)d_in[1];
  const float* b1 = (const float*)d_in[2];
  const float* W2 = (const float*)d_in[3];
  const float* b2 = (const float*)d_in[4];
  const float* am = (const float*)d_in[5];
  const float* nm = (const float*)d_in[6];

  char* ws = (char*)d_ws;
  __bf16* W1h = (__bf16*)(ws + OFF_W1H);
  __bf16* W1l = (__bf16*)(ws + OFF_W1L);
  __bf16* W2h = (__bf16*)(ws + OFF_W2H);
  __bf16* W2l = (__bf16*)(ws + OFF_W2L);
  __bf16* amh = (__bf16*)(ws + OFF_AMH);
  __bf16* aml = (__bf16*)(ws + OFF_AML);
  __bf16* nmh = (__bf16*)(ws + OFF_NMH);
  __bf16* nml = (__bf16*)(ws + OFF_NML);
  float*  hBuf = (float*)(ws + OFF_H);
  float*  out = (float*)d_out;

  prep_kernel<<<(NPREP + 255) / 256, 256, 0, stream>>>(
      W1, W2, am, nm, W1h, W1l, W2h, W2l, amh, aml, nmh, nml);
  gemm1_kernel<<<256, 256, 0, stream>>>(x, W1h, W1l, b1, hBuf);
  head_kernel<<<128, 256, 0, stream>>>(hBuf, W2h, W2l, b2, amh, aml, nmh, nml, out);
}
